// EGNN_75204877353244
// MI455X (gfx1250) — matrix operand order repaired and checked by reading, not compiled, not run on hardware
//
#include <hip/hip_runtime.h>
#include <hip/hip_bf16.h>
#include <math.h>

#define N_REAL 10000
#define E_BASE 160000
#define NGR    8
#define IN_F   9
#define FF     128
#define LL     7
#define NN     (N_REAL + NGR)        // 10008
#define NPAD   10016
#define ETOT   (E_BASE + 2*N_REAL)   // 180000
#define ETILES (ETOT/16)             // 11250
#define NTILES (NPAD/16)             // 626
#define EPSV   1e-5f

typedef __attribute__((ext_vector_type(16))) __bf16 v16bf;
typedef __attribute__((ext_vector_type(8)))  float  v8f;
typedef unsigned short u16;
typedef unsigned int   u32;

union Frag { v16bf v; u32 u[8]; u16 h[16]; };

__device__ __forceinline__ u16 f2bf(float f){
  u32 u = __float_as_uint(f);
  u32 r = (u + 0x7FFFu + ((u >> 16) & 1u)) >> 16;
  return (u16)r;
}
__device__ __forceinline__ float bf2f(u16 h){
  return __uint_as_float(((u32)h) << 16);
}
__device__ __forceinline__ float swishf(float x){
  return x / (1.0f + __expf(-x));
}
__device__ __forceinline__ float sigm(float x){
  return 1.0f / (1.0f + __expf(-x));
}

// implicit augmented edge list: e<E_BASE real, then phantom (virtual->real), then (real->virtual)
__device__ __forceinline__ void edge_sd(int e, const int* ei, const int* batch, int& s, int& d){
  if (e < E_BASE)              { s = ei[e];                 d = ei[E_BASE + e]; }
  else if (e < E_BASE + N_REAL){ int i = e - E_BASE;        s = N_REAL + batch[i]; d = i; }
  else                         { int i = e - E_BASE - N_REAL; s = i;             d = N_REAL + batch[i]; }
}

// 16 x K (bf16 in LDS) times packed K x 128 bf16 weights; this wave computes
// 4 N-tiles starting at ntbase (i.e. output columns [ntbase*16, ntbase*16+64)).
// kb loop fully unrolled (KB is compile-time) so accumulators stay pinned.
// A layout per ISA: lane<16 row M=lane, VGPR v holds K pairs (v<4?0:16)+hl*8+(v&3)*2
template<int KB>
__device__ __forceinline__ void gemm16x4(const u16* A, int astride,
                                         const u16* __restrict__ Wp, int ntbase, v8f* acc){
  const int lane = threadIdx.x & 31;
  const int hl = lane >> 4, idx = lane & 15;
  #pragma unroll
  for (int kb = 0; kb < KB; ++kb){
    Frag a;
    #pragma unroll
    for (int v = 0; v < 8; ++v){
      int k0 = kb*32 + ((v < 4) ? 0 : 16) + hl*8 + (v & 3)*2;
      u32 lo = A[idx*astride + k0];
      u32 hi = A[idx*astride + k0 + 1];
      a.u[v] = lo | (hi << 16);
    }
    #pragma unroll
    for (int nt = 0; nt < 4; ++nt){
      Frag b;
      const u32* wp = (const u32*)(Wp + ((size_t)((kb*8 + ntbase + nt)*32 + lane))*16);
      #pragma unroll
      for (int j = 0; j < 8; ++j) b.u[j] = wp[j];
      acc[nt] = __builtin_amdgcn_wmma_f32_16x16x32_bf16(
          false, a.v, false, b.v, (short)0, acc[nt], false, false);
    }
  }
}

// --- pack fp32 KxN(=128) weights into bf16 B-fragment order ---
// B layout: lane<16 -> col n=lane holds K = (v<4?0:16) + hl*8 + (v&3)*2 (+1) within each 32-K block
__global__ void k_pack(const float* __restrict__ W, u16* __restrict__ out, int K, int KB){
  int t = blockIdx.x*blockDim.x + threadIdx.x;
  if (t >= KB*4096) return;
  int j = t & 15, lane = (t >> 4) & 31, nt = (t >> 9) & 7, kb = t >> 12;
  int v = j >> 1, lo = j & 1;
  int k = kb*32 + ((v < 4) ? 0 : 16) + (lane >> 4)*8 + (v & 3)*2 + lo;
  int n = nt*16 + (lane & 15);
  float val = (k < K) ? W[(size_t)k*FF + n] : 0.0f;
  out[t] = f2bf(val);
}

__global__ void k_node_init(const float* pos, const int* tags, float* posb, float* tagm){
  int i = blockIdx.x*blockDim.x + threadIdx.x;
  if (i >= NPAD) return;
  float x=0, y=0, z=0, tm=0;
  if (i < N_REAL){ x = pos[i*3]; y = pos[i*3+1]; z = pos[i*3+2]; tm = (tags[i]==2) ? 1.f : 0.f; }
  else if (i < NN){ tm = 1.f; }   // virtual pos filled by k_gfin
  posb[i*3] = x; posb[i*3+1] = y; posb[i*3+2] = z; tagm[i] = tm;
}

__global__ void k_gacc(const float* pos, const float* atom_map, const int* an, const int* batch,
                       float* gcnt, float* h0sum, float* possum){
  int i = blockIdx.x*blockDim.x + threadIdx.x;
  if (i >= N_REAL) return;
  int b = batch[i];
  atomicAdd(&gcnt[b], 1.0f);
  const float* am = atom_map + (size_t)an[i]*IN_F;
  for (int k = 0; k < IN_F; ++k) atomicAdd(&h0sum[b*IN_F + k], am[k]);
  atomicAdd(&possum[b*3+0], pos[i*3+0]);
  atomicAdd(&possum[b*3+1], pos[i*3+1]);
  atomicAdd(&possum[b*3+2], pos[i*3+2]);
}

__global__ void k_gfin(const float* gcnt, const float* h0sum, const float* possum,
                       float* h0v, float* posb, float* gca){
  int g = threadIdx.x;
  if (g >= NGR) return;
  float c = fmaxf(gcnt[g], 1.0f);
  for (int k = 0; k < IN_F; ++k) h0v[g*IN_F + k] = h0sum[g*IN_F + k] / c;
  posb[(N_REAL+g)*3+0] = possum[g*3+0] / c;
  posb[(N_REAL+g)*3+1] = possum[g*3+1] / c;
  posb[(N_REAL+g)*3+2] = possum[g*3+2] / c;
  gca[g] = gcnt[g] + 1.0f;
}

__global__ void k_embed(const float* atom_map, const int* an, const float* h0v,
                        const float* Wemb, const float* bemb, float* h){
  int t = blockIdx.x*blockDim.x + threadIdx.x;
  if (t >= NPAD*FF) return;
  int i = t / FF, c = t % FF;
  float acc = 0.f;
  if (i < NN){
    const float* h0 = (i < N_REAL) ? (atom_map + (size_t)an[i]*IN_F) : (h0v + (i - N_REAL)*IN_F);
    acc = bemb[c];
    for (int k = 0; k < IN_F; ++k) acc += h0[k]*Wemb[k*FF + c];
  }
  h[t] = acc;
}

__global__ void k_deg(const int* ei, const int* batch, float* deg){
  int e = blockIdx.x*blockDim.x + threadIdx.x;
  if (e >= ETOT) return;
  int s, d; edge_sd(e, ei, batch, s, d);
  atomicAdd(&deg[d], 1.0f);
}

// --- layer kernel 1: z = swish([x_i,x_j,d2] @ Wm1 + bm1), store bf16 z + BN stats ---
// 64 threads: wave0 -> output cols 0..63, wave1 -> cols 64..127 of one 16-edge tile
__global__ void __launch_bounds__(64)
k_edge1(const float* __restrict__ h, const float* __restrict__ posb,
        const float* __restrict__ cell_off,
        const int* __restrict__ ei, const int* __restrict__ batch,
        const u16* __restrict__ Wm1p, const float* __restrict__ bm1,
        u16* __restrict__ zbuf, float* ssum, float* ssq){
  __shared__ u16 sA[16][288];
  __shared__ int sS[16], sD[16];
  __shared__ float sD2[16];
  int tid  = threadIdx.x;
  int wave = tid >> 5;
  int lane = tid & 31;
  int e0 = blockIdx.x*16;
  if (tid < 16){
    int e = e0 + tid;
    int s, d; edge_sd(e, ei, batch, s, d);
    float ox=0, oy=0, oz=0;
    if (e < E_BASE){ ox = cell_off[e*3]; oy = cell_off[e*3+1]; oz = cell_off[e*3+2]; }
    float dx = posb[d*3]   - posb[s*3]   + ox;
    float dy = posb[d*3+1] - posb[s*3+1] + oy;
    float dz = posb[d*3+2] - posb[s*3+2] + oz;
    sS[tid] = s; sD[tid] = d; sD2[tid] = dx*dx + dy*dy + dz*dz;
  }
  __syncthreads();
  for (int i = tid; i < 16*288; i += 64){
    int r = i / 288, c = i % 288;
    float v;
    if      (c < 128)  v = h[(size_t)sD[r]*FF + c];
    else if (c < 256)  v = h[(size_t)sS[r]*FF + (c - 128)];
    else if (c == 256) v = sD2[r];
    else               v = 0.f;
    sA[r][c] = f2bf(v);
  }
  __syncthreads();
  v8f acc[4];
  const v8f vz = {0,0,0,0,0,0,0,0};
  #pragma unroll
  for (int nt = 0; nt < 4; ++nt) acc[nt] = vz;
  gemm16x4<9>(&sA[0][0], 288, Wm1p, wave*4, acc);
  int hl = lane >> 4, idx = lane & 15;
  #pragma unroll
  for (int nt = 0; nt < 4; ++nt){
    int col = (wave*4 + nt)*16 + idx;
    float bb = bm1[col];
    float ps = 0.f, ps2 = 0.f;
    #pragma unroll
    for (int v = 0; v < 8; ++v){
      int row = hl*8 + v;
      float z = swishf(acc[nt][v] + bb);
      zbuf[(size_t)(e0 + row)*FF + col] = f2bf(z);
      ps += z; ps2 += z*z;
    }
    atomicAdd(&ssum[col], ps);
    atomicAdd(&ssq[col], ps2);
  }
}

__global__ void k_bnfin(const float* ssum, const float* ssq, const float* bng, const float* bnb,
                        float* scale, float* shift){
  int c = threadIdx.x;
  if (c >= FF) return;
  float inv = 1.0f / (float)ETOT;
  float mu  = ssum[c]*inv;
  float var = ssq[c]*inv - mu*mu;
  float sc  = bng[c]*rsqrtf(var + EPSV);
  scale[c] = sc;
  shift[c] = bnb[c] - mu*sc;
}

// --- layer kernel 2: m/gate/pos-message, scatter agg_m and agg_pm ---
__global__ void __launch_bounds__(64)
k_edge2(const float* __restrict__ posb, const float* __restrict__ cell_off,
        const int* __restrict__ ei, const int* __restrict__ batch,
        const u16* __restrict__ zbuf,
        const float* __restrict__ scale, const float* __restrict__ shift,
        const u16* __restrict__ Wm2p, const float* __restrict__ bm2,
        const float* __restrict__ Winf, const float* __restrict__ binf,
        const u16* __restrict__ Wp1p, const float* __restrict__ bp1,
        const float* __restrict__ Wp2, const float* __restrict__ bp2,
        float* agg_m, float* agg_pm){
  __shared__ u16   sZ[16][128];
  __shared__ float sM[16][128];
  __shared__ u16   sMb[16][128];
  __shared__ float sP[16][128];
  __shared__ float sDv[16][3];
  __shared__ float sG[16];
  __shared__ int   sD[16];
  int tid  = threadIdx.x;
  int wave = tid >> 5;
  int lane = tid & 31;
  int e0 = blockIdx.x*16;
  if (tid < 16){
    int e = e0 + tid;
    int s, d; edge_sd(e, ei, batch, s, d);
    float ox=0, oy=0, oz=0;
    if (e < E_BASE){ ox = cell_off[e*3]; oy = cell_off[e*3+1]; oz = cell_off[e*3+2]; }
    sDv[tid][0] = posb[d*3]   - posb[s*3]   + ox;
    sDv[tid][1] = posb[d*3+1] - posb[s*3+1] + oy;
    sDv[tid][2] = posb[d*3+2] - posb[s*3+2] + oz;
    sD[tid] = d;
  }
  for (int i = tid; i < 2048; i += 64){
    int r = i >> 7, c = i & 127;
    float z = bf2f(zbuf[(size_t)(e0 + r)*FF + c]);
    sZ[r][c] = f2bf(z*scale[c] + shift[c]);
  }
  __syncthreads();
  v8f acc[4];
  const v8f vz = {0,0,0,0,0,0,0,0};
  #pragma unroll
  for (int nt = 0; nt < 4; ++nt) acc[nt] = vz;
  gemm16x4<4>(&sZ[0][0], 128, Wm2p, wave*4, acc);
  int hl = lane >> 4, idx = lane & 15;
  #pragma unroll
  for (int nt = 0; nt < 4; ++nt){
    int col = (wave*4 + nt)*16 + idx; float bb = bm2[col];
    #pragma unroll
    for (int v = 0; v < 8; ++v) sM[hl*8 + v][col] = swishf(acc[nt][v] + bb);
  }
  __syncthreads();
  if (tid < 16){
    float dot = binf[0];
    for (int c = 0; c < 128; ++c) dot += sM[tid][c]*Winf[c];
    sG[tid] = sigm(dot);
  }
  __syncthreads();
  for (int i = tid; i < 2048; i += 64){
    int r = i >> 7, c = i & 127;
    float m = sM[r][c]*sG[r];
    sM[r][c] = m;
    sMb[r][c] = f2bf(m);
  }
  __syncthreads();
  #pragma unroll
  for (int nt = 0; nt < 4; ++nt) acc[nt] = vz;
  gemm16x4<4>(&sMb[0][0], 128, Wp1p, wave*4, acc);
  #pragma unroll
  for (int nt = 0; nt < 4; ++nt){
    int col = (wave*4 + nt)*16 + idx; float bb = bp1[col];
    #pragma unroll
    for (int v = 0; v < 8; ++v) sP[hl*8 + v][col] = swishf(acc[nt][v] + bb);
  }
  __syncthreads();
  if (tid < 16){
    float dot = bp2[0];
    for (int c = 0; c < 128; ++c) dot += sP[tid][c]*Wp2[c];
    int d = sD[tid];
    atomicAdd(&agg_pm[d*3+0], sDv[tid][0]*dot);
    atomicAdd(&agg_pm[d*3+1], sDv[tid][1]*dot);
    atomicAdd(&agg_pm[d*3+2], sDv[tid][2]*dot);
  }
  for (int i = tid; i < 2048; i += 64){
    int r = i >> 7, c = i & 127;
    atomicAdd(&agg_m[(size_t)sD[r]*FF + c], sM[r][c]);
  }
}

// --- node update: pos += agg_pm/deg * tag; h += swish(swish([h,agg_m/deg]Wu1)Wu2); graph stats ---
__global__ void __launch_bounds__(64)
k_node(const int* __restrict__ batch, const float* __restrict__ deg,
       const float* __restrict__ tagm,
       const float* __restrict__ agg_m, const float* __restrict__ agg_pm,
       const u16* __restrict__ Wu1p, const float* __restrict__ bu1,
       const u16* __restrict__ Wu2p, const float* __restrict__ bu2,
       float* h, float* posb, float* gsum, float* gsq){
  __shared__ u16 sF[16][256];
  __shared__ u16 sU[16][128];
  __shared__ float sInv[16];
  int tid  = threadIdx.x;
  int wave = tid >> 5;
  int lane = tid & 31;
  int i0 = blockIdx.x*16;
  if (tid < 16){
    int i = i0 + tid;
    float dg  = fmaxf(deg[i], 1.0f);
    float inv = 1.0f / dg;
    sInv[tid] = inv;
    if (i < NN){
      float tm = tagm[i];
      posb[i*3+0] += agg_pm[i*3+0]*inv*tm;
      posb[i*3+1] += agg_pm[i*3+1]*inv*tm;
      posb[i*3+2] += agg_pm[i*3+2]*inv*tm;
    }
  }
  __syncthreads();
  for (int t = tid; t < 16*256; t += 64){
    int r = t >> 8, c = t & 255;
    int i = i0 + r;
    float v = 0.f;
    if (i < NN) v = (c < 128) ? h[(size_t)i*FF + c] : agg_m[(size_t)i*FF + (c - 128)]*sInv[r];
    sF[r][c] = f2bf(v);
  }
  __syncthreads();
  v8f acc[4];
  const v8f vz = {0,0,0,0,0,0,0,0};
  #pragma unroll
  for (int nt = 0; nt < 4; ++nt) acc[nt] = vz;
  gemm16x4<8>(&sF[0][0], 256, Wu1p, wave*4, acc);
  int hl = lane >> 4, idx = lane & 15;
  #pragma unroll
  for (int nt = 0; nt < 4; ++nt){
    int col = (wave*4 + nt)*16 + idx; float bb = bu1[col];
    #pragma unroll
    for (int v = 0; v < 8; ++v) sU[hl*8 + v][col] = f2bf(swishf(acc[nt][v] + bb));
  }
  __syncthreads();
  #pragma unroll
  for (int nt = 0; nt < 4; ++nt) acc[nt] = vz;
  gemm16x4<4>(&sU[0][0], 128, Wu2p, wave*4, acc);
  #pragma unroll
  for (int nt = 0; nt < 4; ++nt){
    int col = (wave*4 + nt)*16 + idx; float bb = bu2[col];
    #pragma unroll
    for (int v = 0; v < 8; ++v){
      int i = i0 + hl*8 + v;
      if (i < NN){
        float hn = h[(size_t)i*FF + col] + swishf(acc[nt][v] + bb);
        h[(size_t)i*FF + col] = hn;
        int b = (i < N_REAL) ? batch[i] : (i - N_REAL);
        atomicAdd(&gsum[b*FF + col], hn);
        atomicAdd(&gsq [b*FF + col], hn*hn);
      }
    }
  }
}

__global__ void k_gnorm_fin(const float* gsum, const float* gsq, const float* gca,
                            float* gm, float* gi){
  int t = blockIdx.x*blockDim.x + threadIdx.x;
  if (t >= NGR*FF) return;
  int g = t / FF;
  float c  = fmaxf(gca[g], 1.0f);
  float mu = gsum[t] / c;
  float var = gsq[t] / c - mu*mu;
  gm[t] = mu; gi[t] = rsqrtf(var + EPSV);
}

__global__ void k_gnorm_apply(const int* batch, const float* gm, const float* gi, float* h){
  int t = blockIdx.x*blockDim.x + threadIdx.x;
  if (t >= NN*FF) return;
  int i = t / FF, c = t % FF;
  int b = (i < N_REAL) ? batch[i] : (i - N_REAL);
  h[t] = (h[t] - gm[b*FF + c])*gi[b*FF + c];
}

__global__ void k_out(const float* h, const float* We1, const float* be1,
                      const float* We2, const float* be2,
                      const float* We3, const float* be3, float* out){
  __shared__ float hl_[NGR][FF];
  __shared__ float e1[NGR][512];
  __shared__ float e2[NGR][FF];
  int t = threadIdx.x;  // 512 threads
  for (int i = t; i < NGR*FF; i += 512){ int g = i/FF, c = i%FF; hl_[g][c] = h[(size_t)(N_REAL+g)*FF + c]; }
  __syncthreads();
  for (int i = t; i < NGR*512; i += 512){
    int g = i/512, o = i%512;
    float a = be1[o];
    for (int k = 0; k < FF; ++k) a += hl_[g][k]*We1[k*512 + o];
    e1[g][o] = swishf(a);
  }
  __syncthreads();
  for (int i = t; i < NGR*FF; i += 512){
    int g = i/FF, o = i%FF;
    float a = be2[o];
    for (int k = 0; k < 512; ++k) a += e1[g][k]*We2[k*FF + o];
    e2[g][o] = swishf(a);
  }
  __syncthreads();
  if (t < NGR){
    float a = be3[0];
    for (int k = 0; k < FF; ++k) a += e2[t][k]*We3[k];
    out[t] = a;
  }
}

extern "C" void kernel_launch(void* const* d_in, const int* in_sizes, int n_in,
                              void* d_out, int out_size, void* d_ws, size_t ws_size,
                              hipStream_t stream){
  (void)in_sizes; (void)n_in; (void)out_size; (void)ws_size;
  const float* pos      = (const float*)d_in[0];
  const float* cell_off = (const float*)d_in[1];
  const float* atom_map = (const float*)d_in[2];
  const float* W_emb = (const float*)d_in[3];
  const float* b_emb = (const float*)d_in[4];
  const float* Wm1 = (const float*)d_in[5];
  const float* bm1 = (const float*)d_in[6];
  const float* bng = (const float*)d_in[7];
  const float* bnb = (const float*)d_in[8];
  const float* Wm2 = (const float*)d_in[9];
  const float* bm2 = (const float*)d_in[10];
  const float* Winf = (const float*)d_in[11];
  const float* binf = (const float*)d_in[12];
  const float* Wp1 = (const float*)d_in[13];
  const float* bp1 = (const float*)d_in[14];
  const float* Wp2 = (const float*)d_in[15];
  const float* bp2 = (const float*)d_in[16];
  const float* Wu1 = (const float*)d_in[17];
  const float* bu1 = (const float*)d_in[18];
  const float* Wu2 = (const float*)d_in[19];
  const float* bu2 = (const float*)d_in[20];
  const float* We1 = (const float*)d_in[21];
  const float* be1 = (const float*)d_in[22];
  const float* We2 = (const float*)d_in[23];
  const float* be2 = (const float*)d_in[24];
  const float* We3 = (const float*)d_in[25];
  const float* be3 = (const float*)d_in[26];
  const int* an    = (const int*)d_in[27];
  const int* ei    = (const int*)d_in[28];
  const int* tags  = (const int*)d_in[29];
  const int* batch = (const int*)d_in[30];
  float* out = (float*)d_out;

  char* w = (char*)d_ws;
  size_t off = 0;
  auto alloc = [&](size_t bytes)->void*{
    void* p = w + off; off = (off + bytes + 255) & ~(size_t)255; return p;
  };
  u16* pWm1 = (u16*)alloc(sizeof(u16)*(size_t)LL*9*4096);
  u16* pWm2 = (u16*)alloc(sizeof(u16)*(size_t)LL*4*4096);
  u16* pWp1 = (u16*)alloc(sizeof(u16)*(size_t)LL*4*4096);
  u16* pWu1 = (u16*)alloc(sizeof(u16)*(size_t)LL*8*4096);
  u16* pWu2 = (u16*)alloc(sizeof(u16)*(size_t)LL*4*4096);
  float* hbuf = (float*)alloc(sizeof(float)*(size_t)NPAD*FF);
  float* posb = (float*)alloc(sizeof(float)*(size_t)NPAD*3);
  float* tagm = (float*)alloc(sizeof(float)*NPAD);
  float* deg  = (float*)alloc(sizeof(float)*NPAD);
  float* aggm = (float*)alloc(sizeof(float)*(size_t)NPAD*FF);
  float* aggp = (float*)alloc(sizeof(float)*(size_t)NPAD*3);
  u16*   zbuf = (u16*)alloc(sizeof(u16)*(size_t)ETOT*FF);
  float* ssum = (float*)alloc(sizeof(float)*FF);
  float* ssq  = (float*)alloc(sizeof(float)*FF);
  float* bnsc = (float*)alloc(sizeof(float)*FF);
  float* bnsh = (float*)alloc(sizeof(float)*FF);
  float* gcnt = (float*)alloc(sizeof(float)*NGR);
  float* h0sum= (float*)alloc(sizeof(float)*NGR*IN_F);
  float* possum=(float*)alloc(sizeof(float)*NGR*3);
  float* h0v  = (float*)alloc(sizeof(float)*NGR*IN_F);
  float* gca  = (float*)alloc(sizeof(float)*NGR);
  float* gsum = (float*)alloc(sizeof(float)*NGR*FF);
  float* gsq  = (float*)alloc(sizeof(float)*NGR*FF);
  float* gmn  = (float*)alloc(sizeof(float)*NGR*FF);
  float* gin  = (float*)alloc(sizeof(float)*NGR*FF);

  // pack all layer weights into WMMA fragment order (bf16)
  for (int l = 0; l < LL; ++l){
    k_pack<<<(9*4096+255)/256,256,0,stream>>>(Wm1 + (size_t)l*257*FF, pWm1 + (size_t)l*9*4096, 257, 9);
    k_pack<<<(4*4096+255)/256,256,0,stream>>>(Wm2 + (size_t)l*FF*FF,  pWm2 + (size_t)l*4*4096, 128, 4);
    k_pack<<<(4*4096+255)/256,256,0,stream>>>(Wp1 + (size_t)l*FF*FF,  pWp1 + (size_t)l*4*4096, 128, 4);
    k_pack<<<(8*4096+255)/256,256,0,stream>>>(Wu1 + (size_t)l*2*FF*FF,pWu1 + (size_t)l*8*4096, 256, 8);
    k_pack<<<(4*4096+255)/256,256,0,stream>>>(Wu2 + (size_t)l*FF*FF,  pWu2 + (size_t)l*4*4096, 128, 4);
  }

  hipMemsetAsync(gcnt,  0, sizeof(float)*NGR,        stream);
  hipMemsetAsync(h0sum, 0, sizeof(float)*NGR*IN_F,   stream);
  hipMemsetAsync(possum,0, sizeof(float)*NGR*3,      stream);
  hipMemsetAsync(deg,   0, sizeof(float)*NPAD,       stream);
  k_node_init<<<(NPAD+255)/256,256,0,stream>>>(pos, tags, posb, tagm);
  k_gacc<<<(N_REAL+255)/256,256,0,stream>>>(pos, atom_map, an, batch, gcnt, h0sum, possum);
  k_gfin<<<1,32,0,stream>>>(gcnt, h0sum, possum, h0v, posb, gca);
  k_embed<<<(NPAD*FF+255)/256,256,0,stream>>>(atom_map, an, h0v, W_emb, b_emb, hbuf);
  k_deg<<<(ETOT+255)/256,256,0,stream>>>(ei, batch, deg);

  for (int l = 0; l < LL; ++l){
    hipMemsetAsync(ssum, 0, sizeof(float)*FF,               stream);
    hipMemsetAsync(ssq,  0, sizeof(float)*FF,               stream);
    hipMemsetAsync(aggm, 0, sizeof(float)*(size_t)NPAD*FF,  stream);
    hipMemsetAsync(aggp, 0, sizeof(float)*(size_t)NPAD*3,   stream);
    hipMemsetAsync(gsum, 0, sizeof(float)*NGR*FF,           stream);
    hipMemsetAsync(gsq,  0, sizeof(float)*NGR*FF,           stream);
    k_edge1<<<ETILES,64,0,stream>>>(hbuf, posb, cell_off, ei, batch,
        pWm1 + (size_t)l*9*4096, bm1 + l*FF, zbuf, ssum, ssq);
    k_bnfin<<<1,128,0,stream>>>(ssum, ssq, bng + l*FF, bnb + l*FF, bnsc, bnsh);
    k_edge2<<<ETILES,64,0,stream>>>(posb, cell_off, ei, batch, zbuf, bnsc, bnsh,
        pWm2 + (size_t)l*4*4096, bm2 + l*FF, Winf + l*FF, binf + l,
        pWp1 + (size_t)l*4*4096, bp1 + l*FF, Wp2 + l*FF, bp2 + l, aggm, aggp);
    k_node<<<NTILES,64,0,stream>>>(batch, deg, tagm, aggm, aggp,
        pWu1 + (size_t)l*8*4096, bu1 + l*FF, pWu2 + (size_t)l*4*4096, bu2 + l*FF,
        hbuf, posb, gsum, gsq);
    k_gnorm_fin<<<(NGR*FF+255)/256,256,0,stream>>>(gsum, gsq, gca, gmn, gin);
    k_gnorm_apply<<<(NN*FF+255)/256,256,0,stream>>>(batch, gmn, gin, hbuf);
  }
  k_out<<<1,512,0,stream>>>(hbuf, We1, be1, We2, be2, We3, be3, out);
}
